// AttnMixBlock_21071109554242
// MI455X (gfx1250) — matrix operand order repaired and checked by reading, not compiled, not run on hardware
//
#include <hip/hip_runtime.h>
#include <hip/hip_bf16.h>

// ---------------- problem constants ----------------
constexpr int Bb   = 4;
constexpr int INx  = 4096;
constexpr int Lx   = 4096;   // OUT == L
constexpr int Ax   = 96;
constexpr int Hx   = 3;
constexpr int Dx   = 32;
constexpr int NBx  = 8;
constexpr int KTOP = 32;

typedef __attribute__((ext_vector_type(16))) _Float16     v16h;
typedef __attribute__((ext_vector_type(8)))  float        v8f;
typedef __attribute__((ext_vector_type(4)))  unsigned int u32x4;
typedef __attribute__((ext_vector_type(4)))  int          i32x4;
typedef __attribute__((ext_vector_type(8)))  int          i32x8;

// =====================================================================
// TDM: stage a [rows x rowlen] f32 tile (row-major, stride rowlen) from
// global into LDS via tensor_load_to_lds (D# per ISA 08_async_tensor §8).
// =====================================================================
#if __has_builtin(__builtin_amdgcn_tensor_load_to_lds)
#define HAVE_TDM 1
__device__ __forceinline__ void tdm_load_tile_f32(const float* gsrc, void* lds_dst,
                                                  int rows, int rowlen, int tensor_rows)
{
    const unsigned long long ga = (unsigned long long)(size_t)gsrc;
    const unsigned int      lo  = (unsigned int)(size_t)lds_dst;  // LDS byte offset
    u32x4 g0;
    g0[0] = 1u;                                              // count=1, user mode
    g0[1] = lo;                                              // lds_addr
    g0[2] = (unsigned int)ga;                                // global_addr[31:0]
    g0[3] = (unsigned int)((ga >> 32) & 0x01ffffffULL)       // global_addr[56:32]
          | (2u << 30);                                      // type=2 ("image")
    i32x8 g1;
    g1[0] = (int)(2u << 16);                                 // wg_mask=0, data_size=4B
    g1[1] = (int)(((unsigned)rowlen & 0xffffu) << 16);       // tensor_dim0[15:0] @ [63:48]
    g1[2] = (int)((((unsigned)rowlen >> 16) & 0xffffu)       // tensor_dim0[31:16]
          |  (((unsigned)tensor_rows & 0xffffu) << 16));     // tensor_dim1[15:0]
    g1[3] = (int)((((unsigned)tensor_rows >> 16) & 0xffffu)  // tensor_dim1[31:16]
          |  (((unsigned)rowlen & 0xffffu) << 16));          // tile_dim0 = rowlen
    g1[4] = (int)((unsigned)rows & 0xffffu);                 // tile_dim1 = rows, tile_dim2=0
    g1[5] = (int)(unsigned)rowlen;                           // tensor_dim0_stride[31:0]
    g1[6] = 0;                                               // stride hi, dim1_stride lo
    g1[7] = 0;
    i32x4 z4 = {0, 0, 0, 0};
#if defined(__clang_major__) && (__clang_major__ >= 23)
    i32x8 z8 = {0, 0, 0, 0, 0, 0, 0, 0};
    __builtin_amdgcn_tensor_load_to_lds(g0, g1, z4, z4, z8, 0);
#else
    __builtin_amdgcn_tensor_load_to_lds(g0, g1, z4, z4, 0);
#endif
}
#else
#define HAVE_TDM 0
#endif

// =====================================================================
// Streaming GEMM (A from global): used for v0 = x @ W_pre^T (K=4096,
// A-tile too large for LDS; HBM-bound on W_pre, ~64MB streamed once).
// One wave -> one 16x16 tile. WMMA wave32 layouts as used here (A/B k contiguous, not ISA table order):
//   A 16x32 f16 : lane l -> m=l&15, kb=(l>>4)*16; elem e -> K = kb+e
//   B 32x16 f16 : lane l -> n=l&15, kb=(l>>4)*16; elem e -> K = kb+e
//   C/D 16x16 f32: lane l -> n=l&15, mb=(l>>4)*8; acc[j] -> row mb+j
// =====================================================================
__global__ void gemm_wmma_f16_kernel(const float* __restrict__ A,
                                     const float* __restrict__ W,
                                     const float* __restrict__ bias,
                                     float* __restrict__ Out,
                                     int N, int K, int Mvalid)
{
    const int lane = threadIdx.x & 31;
    const int n0   = blockIdx.x * 16;
    const int m0   = blockIdx.y * 16;

    const int hi  = lane >> 4;
    const int l15 = lane & 15;

    int arow = m0 + l15;
    if (arow >= Mvalid) arow = Mvalid - 1;          // clamp; masked on store
    const float* __restrict__ Aptr = A + (size_t)arow * K + hi * 16;
    const float* __restrict__ Wptr = W + (size_t)(n0 + l15) * K + hi * 16;

    v8f acc = {};
    for (int k0 = 0; k0 < K; k0 += 32) {
        if (k0 + 32 < K) {
            __builtin_prefetch(Wptr + k0 + 32, 0, 1);   // global_prefetch_b8
            __builtin_prefetch(Aptr + k0 + 32, 0, 1);
        }
        const float* ap = Aptr + k0;
        const float* bp = Wptr + k0;
        v16h a, b;
#pragma unroll
        for (int e = 0; e < 8; ++e) {
            a[e]     = (_Float16)ap[e];
            a[e + 8] = (_Float16)ap[8 + e];
        }
#pragma unroll
        for (int e = 0; e < 16; ++e) b[e] = (_Float16)bp[e];
        acc = __builtin_amdgcn_wmma_f32_16x16x32_f16(false, a, false, b,
                                                     (short)0, acc, false, false);
    }

    const int ncol = n0 + l15;
    const float bv = bias ? bias[ncol] : 0.0f;
#pragma unroll
    for (int j = 0; j < 8; ++j) {
        const int m = m0 + hi * 8 + j;
        if (m >= Mvalid) continue;
        Out[(size_t)m * N + ncol] = acc[j] + bv;
    }
}

// =====================================================================
// LDS-staged GEMM for the token matmuls (K in {96,384}, M multiple of 16).
// Block = 192 threads (6 waves) = one 16-row M tile; TDM stages the A tile
// into LDS once, waves stride the N tiles reading A from LDS.
// =====================================================================
__global__ void gemm_wmma_lds_kernel(const float* __restrict__ A,   // [M][K]
                                     const float* __restrict__ W,   // [N][K]
                                     const float* __restrict__ bias,
                                     const float* __restrict__ resid,
                                     float* __restrict__ Out,
                                     int M, int N, int K, int do_gelu)
{
    __shared__ float As[16 * 384];                 // 24KB max (K<=384)
    const int m0 = blockIdx.x * 16;

#if HAVE_TDM
    if (threadIdx.x < 32) {                        // wave 0 issues the DMA
        tdm_load_tile_f32(A + (size_t)m0 * K, As, 16, K, M);
        __builtin_amdgcn_s_wait_tensorcnt(0);      // s_wait_tensorcnt 0
    }
#else
    for (int i = threadIdx.x; i < 16 * K; i += blockDim.x)
        As[i] = A[(size_t)m0 * K + i];
#endif
    __syncthreads();

    const int lane   = threadIdx.x & 31;
    const int wave   = threadIdx.x >> 5;           // 0..5
    const int nwaves = blockDim.x >> 5;            // 6
    const int hi  = lane >> 4;
    const int l15 = lane & 15;
    const int Ntiles = N >> 4;

    const float* Ap = As + l15 * K + hi * 16;      // LDS row for this lane

    for (int nt = wave; nt < Ntiles; nt += nwaves) {
        const int n0 = nt * 16;
        const float* Wp = W + (size_t)(n0 + l15) * K + hi * 16;
        v8f acc = {};
#pragma unroll 3
        for (int k0 = 0; k0 < K; k0 += 32) {
            const float* ap = Ap + k0;
            const float* bp = Wp + k0;
            v16h a, b;
#pragma unroll
            for (int e = 0; e < 8; ++e) {
                a[e]     = (_Float16)ap[e];        // ds_load
                a[e + 8] = (_Float16)ap[8 + e];
            }
#pragma unroll
            for (int e = 0; e < 16; ++e) b[e] = (_Float16)bp[e];
            acc = __builtin_amdgcn_wmma_f32_16x16x32_f16(false, a, false, b,
                                                         (short)0, acc, false, false);
        }
        const int ncol = n0 + l15;
        const float bv = bias ? bias[ncol] : 0.0f;
#pragma unroll
        for (int j = 0; j < 8; ++j) {
            const int m = m0 + hi * 8 + j;
            float r = acc[j] + bv;
            if (do_gelu) r = 0.5f * r * (1.0f + erff(r * 0.70710678118654752f));
            if (resid) r += resid[(size_t)m * N + ncol];
            Out[(size_t)m * N + ncol] = r;
        }
    }
}

// ---------------- tok = v0[...,None]*w_emb + b_emb ----------------
__global__ void token_embed_kernel(const float* __restrict__ v0,
                                   const float* __restrict__ w_emb,
                                   const float* __restrict__ b_emb,
                                   float* __restrict__ tok, int total)
{
    int i = blockIdx.x * blockDim.x + threadIdx.x;
    if (i >= total) return;
    const int a = i % Ax;
    tok[i] = v0[i / Ax] * w_emb[a] + b_emb[a];
}

// ---------------- LayerNorm over A=96, one wave per token ----------------
__global__ void ln96_kernel(const float* __restrict__ in,
                            const float* __restrict__ g,
                            const float* __restrict__ b,
                            float* __restrict__ out)
{
    const int t    = blockIdx.x;
    const int lane = threadIdx.x;
    const float* row = in + (size_t)t * Ax;
    float x0 = row[lane], x1 = row[lane + 32], x2 = row[lane + 64];
    float s = x0 + x1 + x2;
#pragma unroll
    for (int off = 16; off; off >>= 1) s += __shfl_xor(s, off, 32);
    const float mean = s * (1.0f / 96.0f);
    float d0 = x0 - mean, d1 = x1 - mean, d2 = x2 - mean;
    float vs = d0 * d0 + d1 * d1 + d2 * d2;
#pragma unroll
    for (int off = 16; off; off >>= 1) vs += __shfl_xor(vs, off, 32);
    const float inv = rsqrtf(vs * (1.0f / 96.0f) + 1e-5f);
    float* o = out + (size_t)t * Ax;
    o[lane]      = d0 * inv * g[lane]      + b[lane];
    o[lane + 32] = d1 * inv * g[lane + 32] + b[lane + 32];
    o[lane + 64] = d2 * inv * g[lane + 64] + b[lane + 64];
}

// ---------------- u = head @ w1 (reduce over D=32) -----------------------
__global__ void proj_u_kernel(const float* __restrict__ qk,   // [B*L][A]
                              const float* __restrict__ w1,   // [32]
                              float* __restrict__ u, int BHL)
{
    int i = blockIdx.x * blockDim.x + threadIdx.x;
    if (i >= BHL) return;
    const int l = i % Lx;
    const int h = (i / Lx) % Hx;
    const int b = i / (Lx * Hx);
    const float* row = qk + ((size_t)(b * Lx + l)) * Ax + h * Dx;
    float s = 0.0f;
#pragma unroll
    for (int d = 0; d < Dx; ++d) s += row[d] * w1[d];
    u[i] = s;
}

// ---------------- tau = softplus(spline(h1 @ tau_u)) + 0.05 --------------
__global__ void tau_kernel(const float* __restrict__ h1,
                           const float* __restrict__ tau_u,
                           const float* __restrict__ tau_coeff,
                           float* __restrict__ tau, int BL)
{
    int i = blockIdx.x * blockDim.x + threadIdx.x;
    if (i >= BL) return;
    const float* row = h1 + (size_t)i * Ax;
    float s = 0.0f;
    for (int a = 0; a < Ax; ++a) s += row[a] * tau_u[a];
    const float invd = 1.0f / (6.0f / 7.0f + 1e-6f);
    float f = 0.0f;
#pragma unroll
    for (int j = 0; j < NBx; ++j) {
        const float c = -3.0f + j * (6.0f / 7.0f);
        float hat = 1.0f - fabsf(s - c) * invd;
        f += fmaxf(hat, 0.0f) * tau_coeff[j];
    }
    const float sp = (f > 20.0f) ? f : log1pf(expf(f));
    tau[i] = sp + 0.05f;
}

// ---------------- streaming top-32 of spline kernel scores ---------------
__global__ void topk_kernel(const float* __restrict__ uq,
                            const float* __restrict__ uk,
                            const float* __restrict__ tau,
                            const float* __restrict__ kcoeff,
                            float* __restrict__ tv,
                            int*   __restrict__ ti, int BHL)
{
    int i = blockIdx.x * blockDim.x + threadIdx.x;
    if (i >= BHL) return;
    const int l = i % Lx;
    const int h = (i / Lx) % Hx;
    const int b = i / (Lx * Hx);

    const float qv     = uq[i];
    const float invtau = 1.0f / (tau[b * Lx + l] + 1e-6f);
    const float invd   = 1.0f / (6.0f / 7.0f + 1e-6f);
    float c[NBx];
#pragma unroll
    for (int j = 0; j < NBx; ++j) c[j] = kcoeff[h * NBx + j];

    const float* ukrow = uk + (size_t)(b * Hx + h) * Lx;

    float vals[KTOP];
    int   idxs[KTOP];
#pragma unroll
    for (int j = 0; j < KTOP; ++j) { vals[j] = -3.402823466e38f; idxs[j] = 0; }
    float vmin = -3.402823466e38f;
    int   pmin = 0;

    for (int jk = 0; jk < Lx; ++jk) {
        const float u = qv - ukrow[jk];
        float Kv = 0.0f;
#pragma unroll
        for (int t = 0; t < NBx; ++t) {
            const float ctr = -3.0f + t * (6.0f / 7.0f);
            float hat = 1.0f - fabsf(u - ctr) * invd;
            Kv += fmaxf(hat, 0.0f) * c[t];
        }
        Kv *= invtau;
        if (Kv > vmin) {
            vals[pmin] = Kv; idxs[pmin] = jk;
            vmin = vals[0]; pmin = 0;
#pragma unroll
            for (int t = 1; t < KTOP; ++t)
                if (vals[t] < vmin) { vmin = vals[t]; pmin = t; }
        }
    }
    float* tvp = tv + (size_t)i * KTOP;
    int*   tip = ti + (size_t)i * KTOP;
#pragma unroll
    for (int t = 0; t < KTOP; ++t) { tvp[t] = vals[t]; tip[t] = idxs[t]; }
}

// ---------------- softmax(32) + gathered weighted sum -> ctx -------------
__global__ void ctx_kernel(const float* __restrict__ tv,
                           const int*   __restrict__ ti,
                           const float* __restrict__ v,    // [B*L][A]
                           float* __restrict__ ctx)
{
    const int i    = blockIdx.x;
    const int lane = threadIdx.x;
    const int l = i % Lx;
    const int h = (i / Lx) % Hx;
    const int b = i / (Lx * Hx);

    float val = tv[(size_t)i * KTOP + lane];
    int   idx = ti[(size_t)i * KTOP + lane];

    float m = val;
#pragma unroll
    for (int off = 16; off; off >>= 1) m = fmaxf(m, __shfl_xor(m, off, 32));
    float e = expf(val - m);
    float s = e;
#pragma unroll
    for (int off = 16; off; off >>= 1) s += __shfl_xor(s, off, 32);
    const float attn = e / s;

    __shared__ float sa[KTOP];
    __shared__ int   si[KTOP];
    sa[lane] = attn; si[lane] = idx;
    __syncthreads();

    float acc = 0.0f;
    const float* vbase = v + (size_t)b * Lx * Ax + h * Dx + lane;
#pragma unroll 4
    for (int t = 0; t < KTOP; ++t)
        acc += sa[t] * vbase[(size_t)si[t] * Ax];

    ctx[((size_t)(b * Lx + l)) * Ax + h * Dx + lane] = acc;
}

// ---------------- v2 = y @ w_po + b_po -----------------------------------
__global__ void po_kernel(const float* __restrict__ y,
                          const float* __restrict__ w_po,
                          const float* __restrict__ b_po,
                          float* __restrict__ v2, int BL)
{
    int i = blockIdx.x * blockDim.x + threadIdx.x;
    if (i >= BL) return;
    const float* row = y + (size_t)i * Ax;
    float s = 0.0f;
    for (int a = 0; a < Ax; ++a) s += row[a] * w_po[a];
    v2[i] = s + b_po[0];
}

// ---------------- final LN over 4096, one block per batch row ------------
__global__ void final_ln_kernel(const float* __restrict__ x,
                                const float* __restrict__ v2,
                                const float* __restrict__ g,
                                const float* __restrict__ bb,
                                float* __restrict__ out)
{
    const int b   = blockIdx.x;
    const int tid = threadIdx.x;
    __shared__ float buf[Lx];
    __shared__ float part[8];

    float s = 0.0f;
    for (int j = tid; j < Lx; j += 256) {
        const float r = x[b * Lx + j] + v2[b * Lx + j];
        buf[j] = r;
        s += r;
    }
#pragma unroll
    for (int off = 16; off; off >>= 1) s += __shfl_xor(s, off, 32);
    if ((tid & 31) == 0) part[tid >> 5] = s;
    __syncthreads();
    float tot = 0.0f;
#pragma unroll
    for (int w = 0; w < 8; ++w) tot += part[w];
    const float mean = tot / (float)Lx;

    float vs = 0.0f;
    for (int j = tid; j < Lx; j += 256) { const float d = buf[j] - mean; vs += d * d; }
#pragma unroll
    for (int off = 16; off; off >>= 1) vs += __shfl_xor(vs, off, 32);
    __syncthreads();
    if ((tid & 31) == 0) part[tid >> 5] = vs;
    __syncthreads();
    float vtot = 0.0f;
#pragma unroll
    for (int w = 0; w < 8; ++w) vtot += part[w];
    const float inv = rsqrtf(vtot / (float)Lx + 1e-5f);

    for (int j = tid; j < Lx; j += 256)
        out[b * Lx + j] = (buf[j] - mean) * inv * g[j] + bb[j];
}

// =====================================================================
extern "C" void kernel_launch(void* const* d_in, const int* in_sizes, int n_in,
                              void* d_out, int out_size, void* d_ws, size_t ws_size,
                              hipStream_t stream)
{
    (void)in_sizes; (void)n_in; (void)out_size; (void)ws_size;

    const float* x       = (const float*)d_in[0];
    const float* W_pre   = (const float*)d_in[1];
    const float* b_pre   = (const float*)d_in[2];
    const float* w_emb   = (const float*)d_in[3];
    const float* b_emb   = (const float*)d_in[4];
    const float* ln1_g   = (const float*)d_in[5];
    const float* ln1_b   = (const float*)d_in[6];
    const float* Wq      = (const float*)d_in[7];
    const float* Wk      = (const float*)d_in[8];
    const float* Wv      = (const float*)d_in[9];
    const float* wq1     = (const float*)d_in[10];
    const float* wk1     = (const float*)d_in[11];
    const float* kcoeff  = (const float*)d_in[12];
    const float* tau_u   = (const float*)d_in[13];
    const float* tau_cf  = (const float*)d_in[14];
    const float* Wout    = (const float*)d_in[15];
    const float* ln2_g   = (const float*)d_in[16];
    const float* ln2_b   = (const float*)d_in[17];
    const float* ffn_w1  = (const float*)d_in[18];
    const float* ffn_b1  = (const float*)d_in[19];
    const float* ffn_w2  = (const float*)d_in[20];
    const float* ffn_b2  = (const float*)d_in[21];
    const float* w_po    = (const float*)d_in[22];
    const float* b_po    = (const float*)d_in[23];
    const float* lnf_g   = (const float*)d_in[24];
    const float* lnf_b   = (const float*)d_in[25];
    float* out = (float*)d_out;

    const int BL  = Bb * Lx;         // 16384 tokens
    const int BHL = Bb * Hx * Lx;    // 49152 query rows

    // -------- workspace carving (floats) --------
    float* ws = (float*)d_ws;
    size_t off = 0;
    auto alloc = [&](size_t n) { float* p = ws + off; off += n; return p; };
    float* v0   = alloc((size_t)BL);
    float* tok  = alloc((size_t)BL * Ax);
    float* h1   = alloc((size_t)BL * Ax);
    float* q    = alloc((size_t)BL * Ax);
    float* k    = alloc((size_t)BL * Ax);
    float* v    = alloc((size_t)BL * Ax);
    float* uq   = alloc((size_t)BHL);
    float* uk   = alloc((size_t)BHL);
    float* tau  = alloc((size_t)BL);
    float* tv   = alloc((size_t)BHL * KTOP);
    int*   ti   = (int*)alloc((size_t)BHL * KTOP);
    float* ctx  = alloc((size_t)BL * Ax);
    float* y    = alloc((size_t)BL * Ax);
    float* h2   = alloc((size_t)BL * Ax);
    float* ff1  = alloc((size_t)BL * 4 * Ax);
    float* y2   = alloc((size_t)BL * Ax);
    float* v2   = alloc((size_t)BL);

    const dim3 wave(32);
    const dim3 tgemm(192);                   // 6 waves
    const int  mblocks = BL / 16;            // 1024

    // 1) v0 = x @ W_pre^T + b_pre   (streaming; HBM-bound on W_pre)
    gemm_wmma_f16_kernel<<<dim3(Lx / 16, 1), wave, 0, stream>>>(
        x, W_pre, b_pre, v0, /*N=*/Lx, /*K=*/INx, /*Mvalid=*/Bb);

    // 2) tok, h1 = LN1(tok)
    {
        const int total = BL * Ax;
        token_embed_kernel<<<(total + 255) / 256, 256, 0, stream>>>(v0, w_emb, b_emb, tok, total);
        ln96_kernel<<<BL, wave, 0, stream>>>(tok, ln1_g, ln1_b, h1);
    }

    // 3) q, k, v projections (TDM-staged A tile + WMMA)
    gemm_wmma_lds_kernel<<<mblocks, tgemm, 0, stream>>>(h1, Wq, nullptr, nullptr, q, BL, Ax, Ax, 0);
    gemm_wmma_lds_kernel<<<mblocks, tgemm, 0, stream>>>(h1, Wk, nullptr, nullptr, k, BL, Ax, Ax, 0);
    gemm_wmma_lds_kernel<<<mblocks, tgemm, 0, stream>>>(h1, Wv, nullptr, nullptr, v, BL, Ax, Ax, 0);

    // 4) uq, uk, tau
    proj_u_kernel<<<(BHL + 255) / 256, 256, 0, stream>>>(q, wq1, uq, BHL);
    proj_u_kernel<<<(BHL + 255) / 256, 256, 0, stream>>>(k, wk1, uk, BHL);
    tau_kernel<<<(BL + 255) / 256, 256, 0, stream>>>(h1, tau_u, tau_cf, tau, BL);

    // 5) streaming top-32 spline attention scores
    topk_kernel<<<(BHL + 255) / 256, 256, 0, stream>>>(uq, uk, tau, kcoeff, tv, ti, BHL);

    // 6) softmax + gather-weighted context
    ctx_kernel<<<BHL, wave, 0, stream>>>(tv, ti, v, ctx);

    // 7) y = tok + ctx @ Wout^T ; h2 = LN2(y)
    gemm_wmma_lds_kernel<<<mblocks, tgemm, 0, stream>>>(ctx, Wout, nullptr, tok, y, BL, Ax, Ax, 0);
    ln96_kernel<<<BL, wave, 0, stream>>>(y, ln2_g, ln2_b, h2);

    // 8) FFN: gelu(h2 @ w1^T + b1) @ w2^T + b2 + y
    gemm_wmma_lds_kernel<<<mblocks, tgemm, 0, stream>>>(h2, ffn_w1, ffn_b1, nullptr, ff1, BL, 4 * Ax, Ax, 1);
    gemm_wmma_lds_kernel<<<mblocks, tgemm, 0, stream>>>(ff1, ffn_w2, ffn_b2, y, y2, BL, Ax, 4 * Ax, 0);

    // 9) v2 = y2 @ w_po + b_po ; out = LN_f(x + v2)
    po_kernel<<<(BL + 255) / 256, 256, 0, stream>>>(y2, w_po, b_po, v2, BL);
    final_ln_kernel<<<Bb, 256, 0, stream>>>(x, v2, lnf_g, lnf_b, out);
}
